// AttentionTenpow2_65524021068295
// MI455X (gfx1250) — matrix operand order repaired and checked by reading, not compiled, not run on hardware
//
#include <hip/hip_runtime.h>

// Poly-2 causal linear attention for MI455X (gfx1250, wave32, WMMA).
// out[t,v] = sum_{m<=t} (Q[t].K[m])^2 V[m,v] / (sum_{m<=t} (Q[t].K[m])^2 + 1e-5)

typedef __attribute__((ext_vector_type(16))) _Float16     v16h;
typedef __attribute__((ext_vector_type(8)))  float        v8f;
typedef __attribute__((ext_vector_type(4)))  unsigned int v4u;

#define TSEQ  2048
#define NHEAD 8
#define DHEAD 64
#define CLEN  128    // chunk length = t / chunk_count
#define NCHK  16     // number of chunks
#define NWAVE 8

union Frag {
  v16h     v;
  _Float16 h[16];
  v4u      q[2];     // two 16-byte halves for ds_load_b128-style moves
};

__global__ __launch_bounds__(256)
void poly2_attn(const float* __restrict__ Qg,
                const float* __restrict__ Kg,
                const float* __restrict__ Vg,
                float* __restrict__ Og)
{
  __shared__ _Float16 sK [CLEN * DHEAD];    // K chunk, [kpos][d], f16, 16 KB
  __shared__ _Float16 sVt[DHEAD * CLEN];    // V chunk transposed, [v][kpos], 16 KB
  __shared__ _Float16 sP [NWAVE][16 * 32];  // per-wave P re-layout staging, 8 KB

  const int qc    = blockIdx.x;             // query chunk index
  const int hh    = blockIdx.y;
  const int bb    = blockIdx.z;
  const int tid   = threadIdx.x;
  const int wave  = tid >> 5;               // 8 waves, 16 query rows each
  const int lane  = tid & 31;
  const int lmod  = lane & 15;
  const int lhalf = lane >> 4;              // 0 / 1
  const int mbase = lhalf * 8;              // accumulator row base for this lane

  // X[b][t][h][d] flat index
  auto gidx = [&](int t, int d) -> size_t {
    return ((((size_t)bb * TSEQ) + t) * NHEAD + hh) * DHEAD + d;
  };

  v8f vzero;
#pragma unroll
  for (int e = 0; e < 8; ++e) vzero[e] = 0.0f;

  // ---- Q A-fragments for this wave's 16 rows (d split 0..31 / 32..63) ----
  // A layout (16x32 f16): lane row = lane%16; elems 0..15 <-> K=16*half+{0..15}
  // contiguous.
  const int qrow0 = qc * CLEN + wave * 16;
  Frag a0, a1;
  {
    const float* qp = Qg + gidx(qrow0 + lmod, 0);
    const int d1 = 16 * lhalf;
    const int d2 = 16 * lhalf + 8;
#pragma unroll
    for (int e = 0; e < 8; ++e) {
      a0.h[e]     = (_Float16)qp[d1 + e];
      a0.h[8 + e] = (_Float16)qp[d2 + e];
      a1.h[e]     = (_Float16)qp[32 + d1 + e];
      a1.h[8 + e] = (_Float16)qp[32 + d2 + e];
    }
  }

  // ---- ones B-fragment: column 0 = 1 -> accumulates row-sum Z via WMMA ----
  Frag ones;
#pragma unroll
  for (int e = 0; e < 16; ++e)
    ones.h[e] = (lmod == 0) ? (_Float16)1.0f : (_Float16)0.0f;

  // acc[0..3] = Y (16x64), acc[4] col0 = Z
  v8f acc[5];
#pragma unroll
  for (int i = 0; i < 5; ++i) acc[i] = vzero;

  for (int kc = 0; kc <= qc; ++kc) {
    __syncthreads();
    // ---- cooperative stage: K chunk (row major) + V chunk (transposed), f32->f16
#pragma unroll
    for (int i = 0; i < 8; ++i) {
      const int e4  = tid + i * 256;        // float4 index within 128x64
      const int row = e4 >> 4;
      const int d4  = (e4 & 15) * 4;
      const float4 kq = *(const float4*)(Kg + gidx(kc * CLEN + row, d4));
      const float4 vq = *(const float4*)(Vg + gidx(kc * CLEN + row, d4));
      sK[row * DHEAD + d4 + 0] = (_Float16)kq.x;
      sK[row * DHEAD + d4 + 1] = (_Float16)kq.y;
      sK[row * DHEAD + d4 + 2] = (_Float16)kq.z;
      sK[row * DHEAD + d4 + 3] = (_Float16)kq.w;
      sVt[(d4 + 0) * CLEN + row] = (_Float16)vq.x;
      sVt[(d4 + 1) * CLEN + row] = (_Float16)vq.y;
      sVt[(d4 + 2) * CLEN + row] = (_Float16)vq.z;
      sVt[(d4 + 3) * CLEN + row] = (_Float16)vq.w;
    }
    __syncthreads();

    const bool diag   = (kc == qc);
    const int ngroups = diag ? ((wave >> 1) + 1) : 4;  // 32-key groups

    for (int g = 0; g < ngroups; ++g) {
      const int kbase = g * 32;

      // ---- two 16x16 P tiles: P = (Q K^T)^2, causal-masked ----
#pragma unroll
      for (int tt = 0; tt < 2; ++tt) {
        const int ti = g * 2 + tt;               // 16-key tile index in chunk
        v8f p = vzero;
        if (!(diag && ti > wave)) {
          // B layout (32x16): lane col = lane%16, K = 16*half + {0..15} contiguous
          Frag b0, b1;
          const int krow = kbase + tt * 16 + lmod;
          const _Float16* kp = &sK[krow * DHEAD + 16 * lhalf];
          b0.q[0] = *(const v4u*)(kp);
          b0.q[1] = *(const v4u*)(kp + 8);
          b1.q[0] = *(const v4u*)(kp + 32);
          b1.q[1] = *(const v4u*)(kp + 40);
          p = __builtin_amdgcn_wmma_f32_16x16x32_f16(false, a0.v, false, b0.v,
                                                     (short)0, p, false, false);
          p = __builtin_amdgcn_wmma_f32_16x16x32_f16(false, a1.v, false, b1.v,
                                                     (short)0, p, false, false);
          p = p * p;                               // score^2 in f32
          if (diag && ti == wave) {                // triangular mask on diagonal tile
#pragma unroll
            for (int r = 0; r < 8; ++r)
              if (lmod > r + mbase) p[r] = 0.0f;
          }
        }
        // scatter C-layout tile into wave-private LDS (f32->f16)
#pragma unroll
        for (int r = 0; r < 8; ++r)
          sP[wave][(r + mbase) * 32 + lmod + 16 * tt] = (_Float16)p[r];
      }
      __builtin_amdgcn_wave_barrier();  // same-wave LDS is in-order; block reordering

      // ---- reload P as a 16x32 A-fragment ----
      Frag pa;
      {
        const _Float16* pp = &sP[wave][lmod * 32 + 8 * lhalf];
        pa.q[0] = *(const v4u*)(pp);        // K = 8*half + {0..7}
        pa.q[1] = *(const v4u*)(pp + 16);   // K = 16 + 8*half + {0..7}
      }

      // ---- Y += P * V  (4 N-tiles of 16), Z += P * ones ----
#pragma unroll
      for (int vt = 0; vt < 4; ++vt) {
        Frag bv;
        const _Float16* vp = &sVt[(vt * 16 + lmod) * CLEN + kbase + 8 * lhalf];
        bv.q[0] = *(const v4u*)(vp);
        bv.q[1] = *(const v4u*)(vp + 16);
        acc[vt] = __builtin_amdgcn_wmma_f32_16x16x32_f16(false, pa.v, false, bv.v,
                                                         (short)0, acc[vt], false, false);
      }
      acc[4] = __builtin_amdgcn_wmma_f32_16x16x32_f16(false, pa.v, false, ones.v,
                                                      (short)0, acc[4], false, false);
      __builtin_amdgcn_wave_barrier();
    }
  }

  // ---- epilogue: broadcast Z (held by lanes 0 / 16), normalize, store ----
#pragma unroll
  for (int r = 0; r < 8; ++r) {
    const float z   = __shfl(acc[4][r], lane & 16, 32);
    const float inv = 1.0f / (z + 1e-5f);
    const int trow  = qrow0 + r + mbase;
#pragma unroll
    for (int vt = 0; vt < 4; ++vt)
      Og[gidx(trow, vt * 16 + lmod)] = acc[vt][r] * inv;
  }
}

extern "C" void kernel_launch(void* const* d_in, const int* in_sizes, int n_in,
                              void* d_out, int out_size, void* d_ws, size_t ws_size,
                              hipStream_t stream) {
  (void)in_sizes; (void)n_in; (void)d_ws; (void)ws_size; (void)out_size;
  const float* Q = (const float*)d_in[0];
  const float* K = (const float*)d_in[1];
  const float* V = (const float*)d_in[2];
  float* O       = (float*)d_out;
  dim3 grid(NCHK, NHEAD, 2);   // (query chunk, head, batch) = 256 workgroups
  dim3 block(256);             // 8 wave32 waves
  hipLaunchKernelGGL(poly2_attn, grid, block, 0, stream, Q, K, V, O);
}
